// MultiExpertClassifier_52484500357369
// MI455X (gfx1250) — matrix operand order repaired and checked by reading, not compiled, not run on hardware
//
#include <hip/hip_runtime.h>
#include <math.h>

typedef _Float16 half8  __attribute__((ext_vector_type(8)));
typedef _Float16 half16 __attribute__((ext_vector_type(16)));
typedef float    float8 __attribute__((ext_vector_type(8)));
typedef unsigned int uint4v __attribute__((ext_vector_type(4)));
typedef int      int8v  __attribute__((ext_vector_type(8)));
typedef int      int4v  __attribute__((ext_vector_type(4)));

#define DDIM   512
#define CDIM   1000
#define CPAD   1024
#define NEXP   3
#define SCALEF 30.0f
#define EPSF   1e-12f
#define WST    520        // padded f16 row stride for weights (bank-friendly in LDS)

#if defined(__has_builtin)
#if __has_builtin(__builtin_amdgcn_tensor_load_to_lds) && \
    __has_builtin(__builtin_amdgcn_s_wait_tensorcnt)
#define HAVE_TDM 1
#endif
#endif

// ---------------------------------------------------------------------------
// Kernel 1: normalize the 3 coarse weight rows (fp32, into ws) + zero counters
// ---------------------------------------------------------------------------
__global__ __launch_bounds__(256) void mec_norm_coarse(
    const float* __restrict__ wc, float* __restrict__ cwn, int* __restrict__ counts) {
  __shared__ float red[256];
  const int e = blockIdx.x, t = threadIdx.x;
  if (t == 0) counts[e] = 0;
  float v0 = wc[e * DDIM + t];
  float v1 = wc[e * DDIM + 256 + t];
  red[t] = v0 * v0 + v1 * v1;
  __syncthreads();
  for (int off = 128; off > 0; off >>= 1) {
    if (t < off) red[t] += red[t + off];
    __syncthreads();
  }
  float inv = 1.0f / fmaxf(sqrtf(red[0]), EPSF);
  cwn[e * DDIM + t]       = v0 * inv;
  cwn[e * DDIM + 256 + t] = v1 * inv;
}

// ---------------------------------------------------------------------------
// Kernel 2: normalize fine expert rows, fold *30, convert to f16, pad C->1024.
// wb layout: [3][1024] rows of WST(=520) f16, row-major (feeds WMMA B + TDM).
// ---------------------------------------------------------------------------
__global__ __launch_bounds__(256) void mec_norm_fine(
    const float* __restrict__ w0, const float* __restrict__ w1,
    const float* __restrict__ w2, _Float16* __restrict__ wb) {
  __shared__ float red[256];
  const int blk = blockIdx.x;
  const int e = blk >> 10;          // /1024
  const int c = blk & (CPAD - 1);
  const int t = threadIdx.x;
  _Float16* dst = wb + (size_t)blk * WST;
  if (c >= CDIM) {                  // zero padding rows (uniform per block)
    dst[t] = (_Float16)0.0f;
    dst[t + 256] = (_Float16)0.0f;
    if (t < WST - DDIM) dst[DDIM + t] = (_Float16)0.0f;
    return;
  }
  const float* w = (e == 0) ? w0 : ((e == 1) ? w1 : w2);
  const float* row = w + (size_t)c * DDIM;
  float v0 = row[t];
  float v1 = row[t + 256];
  red[t] = v0 * v0 + v1 * v1;
  __syncthreads();
  for (int off = 128; off > 0; off >>= 1) {
    if (t < off) red[t] += red[t + off];
    __syncthreads();
  }
  float inv = SCALEF / fmaxf(sqrtf(red[0]), EPSF);
  dst[t]       = (_Float16)(v0 * inv);
  dst[t + 256] = (_Float16)(v1 * inv);
  if (t < WST - DDIM) dst[DDIM + t] = (_Float16)0.0f;  // K padding tail
}

// ---------------------------------------------------------------------------
// Kernel 3: per-token (one wave32 per token): ||x||, coarse logits, argmax,
// write coarse+select outputs, emit normalized f16 row, append to expert list.
// ---------------------------------------------------------------------------
__global__ __launch_bounds__(256) void mec_route(
    const float* __restrict__ x, const float* __restrict__ cwn,
    float* __restrict__ out_coarse, float* __restrict__ out_sel,
    _Float16* __restrict__ xh, int* __restrict__ counts,
    int* __restrict__ lists, int B) {
  const int lane = threadIdx.x & 31;
  const int wv   = threadIdx.x >> 5;
  const int token = blockIdx.x * 8 + wv;
  const float* xr = x + (size_t)token * DDIM;

  float vv[16];
  float ss = 0.f, d0 = 0.f, d1 = 0.f, d2 = 0.f;
#pragma unroll
  for (int k = 0; k < 16; ++k) {
    float v = xr[k * 32 + lane];
    vv[k] = v;
    ss += v * v;
    d0 += v * cwn[0 * DDIM + k * 32 + lane];
    d1 += v * cwn[1 * DDIM + k * 32 + lane];
    d2 += v * cwn[2 * DDIM + k * 32 + lane];
  }
#pragma unroll
  for (int o = 16; o > 0; o >>= 1) {     // xor-reduce: result in all lanes
    ss += __shfl_xor(ss, o, 32);
    d0 += __shfl_xor(d0, o, 32);
    d1 += __shfl_xor(d1, o, 32);
    d2 += __shfl_xor(d2, o, 32);
  }
  float inv = 1.0f / fmaxf(sqrtf(ss), EPSF);
  float c0 = d0 * inv * SCALEF;
  float c1 = d1 * inv * SCALEF;
  float c2 = d2 * inv * SCALEF;
  int sel = 0; float best = c0;
  if (c1 > best) { best = c1; sel = 1; }   // strict > == argmax first-max
  if (c2 > best) { best = c2; sel = 2; }
  if (lane == 0) {
    out_coarse[(size_t)token * 3 + 0] = c0;
    out_coarse[(size_t)token * 3 + 1] = c1;
    out_coarse[(size_t)token * 3 + 2] = c2;
    out_sel[token] = (float)sel;
    int slot = atomicAdd(&counts[sel], 1);
    lists[(size_t)sel * B + slot] = token;
  }
  _Float16* xo = xh + (size_t)token * DDIM;
#pragma unroll
  for (int k = 0; k < 16; ++k)
    xo[k * 32 + lane] = (_Float16)(vv[k] * inv);
}

// ---------------------------------------------------------------------------
// Kernel 4: routed WMMA GEMM, TDM-staged B.
//  grid.x = 3 * (B/64): 4 gathered 16-token m-tiles per block (one per wave)
//  grid.y = 1024/64   : 64-column blocks
// One TENSOR_LOAD_TO_LDS stages the 64x520-f16 weight tile into LDS per block;
// inner loop = ds_load_b128 B frags + global A frags + v_wmma_f32_16x16x32_f16.
// ---------------------------------------------------------------------------
__global__ __launch_bounds__(128) void mec_gemm(
    const _Float16* __restrict__ xh, const _Float16* __restrict__ wb,
    const int* __restrict__ lists, const int* __restrict__ counts,
    float* __restrict__ outp, int B) {
  __shared__ __align__(16) _Float16 lb[64 * WST];   // 66,560 B of the 320 KB WGP LDS

  const int lane = threadIdx.x & 31;
  const int wv   = threadIdx.x >> 5;
  const int m4tiles = B >> 6;                 // 4-m-tile groups per expert
  const int e   = blockIdx.x / m4tiles;
  const int mt4 = blockIdx.x - e * m4tiles;
  const int count = counts[e];
  if (mt4 * 64 >= count) return;              // block-uniform, before barrier
  const int* list = lists + (size_t)e * B;
  const int cb = blockIdx.y * 64;
  const _Float16* gtile = wb + (size_t)(e * CPAD + cb) * WST;

  // ---- stage B tile into LDS --------------------------------------------
#ifdef HAVE_TDM
  if (wv == 0) {                              // one wave drives the TDM
    unsigned lds_off = (unsigned)(uintptr_t)&lb[0];
    unsigned long long ga = (unsigned long long)(uintptr_t)gtile;
    uint4v g0;
    g0[0] = 1u;                               // count=1 (valid), no gather
    g0[1] = lds_off;                          // lds_addr
    g0[2] = (unsigned)(ga & 0xFFFFFFFFu);     // global_addr[31:0]
    g0[3] = (unsigned)((ga >> 32) & 0x01FFFFFFu) | (2u << 30);  // addr[56:32]|type=2
    int8v g1;
    g1[0] = (1 << 16);                        // data_size=2B, no pad/iterate
    g1[1] = (WST << 16);                      // tensor_dim0[15:0] in [31:16]
    g1[2] = (64 << 16);                       // tensor_dim0 hi=0 | tensor_dim1=64
    g1[3] = (WST << 16);                      // tensor_dim1 hi=0 | tile_dim0=520
    g1[4] = 64;                               // tile_dim1=64, tile_dim2=0
    g1[5] = WST;                              // tensor_dim0_stride[31:0]
    g1[6] = 0;
    g1[7] = 0;
    int4v gz = {0, 0, 0, 0};                  // groups 2/3 unused (2-D tensor)
    int8v gz8 = {0, 0, 0, 0, 0, 0, 0, 0};     // unused trailing descriptor group
    __builtin_amdgcn_tensor_load_to_lds(g0, g1, gz, gz, gz8, 0);
    __builtin_amdgcn_s_wait_tensorcnt(0);
  }
  __syncthreads();
#else
  for (int i = threadIdx.x; i < (64 * WST) / 8; i += 128)
    *(half8*)(lb + (size_t)i * 8) = *(const half8*)(gtile + (size_t)i * 8);
  __syncthreads();
#endif

  // ---- per-wave 16x64 WMMA GEMM -----------------------------------------
  const int mt = mt4 * 4 + wv;
  if (mt * 16 < count) {                      // wave-uniform, no barriers inside
    const int hf = lane >> 4;
    const int ln = lane & 15;
    int slot = mt * 16 + ln;
    int gs = slot < count ? slot : mt * 16;   // branchless clamp to valid slot
    const _Float16* ap = xh + (size_t)list[gs] * DDIM;

    float8 acc[4] = {};
    for (int kb = 0; kb < DDIM; kb += 32) {
      // A fragment: row L%16, two contiguous 8-half chunks (ISA 16-bit A layout)
      half8 a0 = *(const half8*)(ap + kb + 8 * hf);
      half8 a1 = *(const half8*)(ap + kb + 8 * hf + 16);
      half16 A = __builtin_shufflevector(a0, a1,
          0,1,2,3,4,5,6,7,8,9,10,11,12,13,14,15);
#pragma unroll
      for (int s = 0; s < 4; ++s) {
        // B fragment from LDS: col L%16, two 8-half chunks at K 8*(L/16) and 8*(L/16)+16
        const _Float16* bp = lb + (size_t)(s * 16 + ln) * WST + kb + 8 * hf;
        half8 b0 = *(const half8*)bp;
        half8 b1 = *(const half8*)(bp + 16);
        half16 Bm = __builtin_shufflevector(b0, b1,
            0,1,2,3,4,5,6,7,8,9,10,11,12,13,14,15);
        acc[s] = __builtin_amdgcn_wmma_f32_16x16x32_f16(
            false, A, false, Bm, (short)0, acc[s], false, false);
      }
    }
    // Scatter 16x16 f32 tiles: VGPR r = row r+8*hf, lanes = 16 consecutive cols
#pragma unroll
    for (int s = 0; s < 4; ++s) {
      int n = cb + s * 16 + ln;
      if (n < CDIM) {
#pragma unroll
        for (int r = 0; r < 8; ++r) {
          int s2 = mt * 16 + 8 * hf + r;
          if (s2 < count) {
            outp[(size_t)list[s2] * CDIM + n] = acc[s][r];
          }
        }
      }
    }
  }
}

// ---------------------------------------------------------------------------
extern "C" void kernel_launch(void* const* d_in, const int* in_sizes, int n_in,
                              void* d_out, int out_size, void* d_ws, size_t ws_size,
                              hipStream_t stream) {
  const float* x  = (const float*)d_in[0];
  const float* wc = (const float*)d_in[1];
  const float* w0 = (const float*)d_in[2];
  const float* w1 = (const float*)d_in[3];
  const float* w2 = (const float*)d_in[4];
  float* out = (float*)d_out;
  const int B = in_sizes[0] / DDIM;      // 65536

  // workspace carve-up (~71.5 MB total)
  char* ws = (char*)d_ws;
  float* cwn    = (float*)ws;                          // 3*512 f32
  int*   counts = (int*)(ws + 8192);                   // 3 ints
  int*   lists  = (int*)(ws + 16384);                  // 3*B ints
  size_t xh_off = 16384 + (size_t)NEXP * B * sizeof(int);
  xh_off = (xh_off + 1023) & ~(size_t)1023;
  _Float16* xh = (_Float16*)(ws + xh_off);             // B*512 f16
  size_t wb_off = xh_off + (size_t)B * DDIM * sizeof(_Float16);
  wb_off = (wb_off + 1023) & ~(size_t)1023;
  _Float16* wb = (_Float16*)(ws + wb_off);             // 3*1024*520 f16

  float* out_coarse = out;                       // [B,3]
  float* out_sel    = out + (size_t)B * 3;       // [B]
  float* out_logits = out + (size_t)B * 4;       // [B,1000]

  mec_norm_coarse<<<NEXP, 256, 0, stream>>>(wc, cwn, counts);
  mec_norm_fine<<<NEXP * CPAD, 256, 0, stream>>>(w0, w1, w2, wb);
  mec_route<<<B / 8, 256, 0, stream>>>(x, cwn, out_coarse, out_sel, xh,
                                       counts, lists, B);
  dim3 g(NEXP * (B / 64), CPAD / 64);
  mec_gemm<<<g, 128, 0, stream>>>(xh, wb, lists, counts, out_logits, B);
}
